// Stage_721554506395
// MI455X (gfx1250) — matrix operand order repaired and checked by reading, not compiled, not run on hardware
//
#include <hip/hip_runtime.h>
#include <hip/hip_bf16.h>

// ---------------- CDNA5 WMMA types ----------------
typedef __attribute__((ext_vector_type(16))) _Float16 v16h;
typedef __attribute__((ext_vector_type(8)))  float    v8f;
typedef __attribute__((ext_vector_type(4)))  unsigned int u32x4;
typedef __attribute__((ext_vector_type(8)))  int     i32x8;
typedef __attribute__((ext_vector_type(4)))  int     i32x4;

#define DD 6
#define HH 64
#define WWID 64
#define TOK 24576          // 6*64*64 tokens
#define CCH 128

__device__ __forceinline__ v8f wmma16(v16h a, v16h b, v8f c) {
  return __builtin_amdgcn_wmma_f32_16x16x32_f16(false, a, false, b, (short)0, c, false, false);
}

// A-fragment (16x32 f16) per-lane K index for half j (0..15)
__device__ __forceinline__ int a_kmap(int j, int laneHi) {
  return j + (laneHi ? 16 : 0);
}

// ---------------- Tensor Data Mover: 2D tile (rows x rowLen f16) -> LDS ----------------
// D# per CDNA5 ISA ch.8: group0 = {count/flags, lds_addr, global_addr, type=2},
// group1 = {data_size=2B, tensor dims, tile dims, strides}. Contiguous 2D copy.
// This toolchain exposes the 6-arg builtin: (g0, g1, g2, g3, g4, cpol).
__device__ __forceinline__ void tdm_load_2d_f16(unsigned lds_off, const void* gptr,
                                                int rowLen, int rows) {
  unsigned long long ga = (unsigned long long)gptr;
  u32x4 g0;
  g0[0] = 1u;                                            // count=1, user mode
  g0[1] = lds_off;                                       // lds_addr (bytes)
  g0[2] = (unsigned)(ga & 0xffffffffu);                  // global_addr[31:0]
  g0[3] = (unsigned)((ga >> 32) & 0x01ffffffu)           // global_addr[56:32]
          | 0x80000000u;                                 // type=2 ("image") in [127:126]
  i32x8 g1;
  g1[0] = 1 << 16;                                       // data_size=1 (2 bytes)
  g1[1] = (rowLen & 0xffff) << 16;                       // tensor_dim0[15:0] @ bits 63:48
  g1[2] = ((rowLen >> 16) & 0xffff)                      // tensor_dim0[31:16]
          | ((rows & 0xffff) << 16);                     // tensor_dim1[15:0]
  g1[3] = ((rows >> 16) & 0xffff)                        // tensor_dim1[31:16]
          | ((rowLen & 0xffff) << 16);                   // tile_dim0 @ bits 127:112
  g1[4] = rows & 0xffff;                                 // tile_dim1 (tile_dim2 = 0)
  g1[5] = rowLen;                                        // tensor_dim0_stride[31:0]
  g1[6] = (rowLen & 0xffff) << 16;                       // stride0 hi=0 | stride1[15:0]
  g1[7] = (rowLen >> 16) & 0xffff;                       // stride1[47:16]
  i32x4 z4 = {};
  i32x8 z8 = {};
  __builtin_amdgcn_tensor_load_to_lds(g0, g1, z4, z4, z8, 0);
}

// ---------------- LayerNorm kernels ----------------
// From channel-major input x[c][t] -> token-major X[t][c], LN over c.
__global__ void k_ln0(const float* __restrict__ x, const float* __restrict__ g,
                      const float* __restrict__ b, float* __restrict__ X) {
  int t = blockIdx.x, c = threadIdx.x;
  float v = x[(size_t)c * TOK + t];
  __shared__ float red[128];
  red[c] = v; __syncthreads();
  for (int s = 64; s > 0; s >>= 1) { if (c < s) red[c] += red[c + s]; __syncthreads(); }
  float mean = red[0] / 128.f; __syncthreads();
  float d = v - mean;
  red[c] = d * d; __syncthreads();
  for (int s = 64; s > 0; s >>= 1) { if (c < s) red[c] += red[c + s]; __syncthreads(); }
  float var = red[0] / 128.f;
  X[(size_t)t * CCH + c] = d * rsqrtf(var + 1e-5f) * g[c] + b[c];
}

// Token-major LN, outputs f16 (for GEMM A operands) and optionally f32.
__global__ void k_ln_tok(const float* __restrict__ X, const float* __restrict__ g,
                         const float* __restrict__ b, _Float16* __restrict__ O16,
                         float* __restrict__ O32) {
  int t = blockIdx.x, c = threadIdx.x;
  float v = X[(size_t)t * CCH + c];
  __shared__ float red[128];
  red[c] = v; __syncthreads();
  for (int s = 64; s > 0; s >>= 1) { if (c < s) red[c] += red[c + s]; __syncthreads(); }
  float mean = red[0] / 128.f; __syncthreads();
  float d = v - mean;
  red[c] = d * d; __syncthreads();
  for (int s = 64; s > 0; s >>= 1) { if (c < s) red[c] += red[c + s]; __syncthreads(); }
  float var = red[0] / 128.f;
  float y = d * rsqrtf(var + 1e-5f) * g[c] + b[c];
  if (O16) O16[(size_t)t * CCH + c] = (_Float16)y;
  if (O32) O32[(size_t)t * CCH + c] = y;
}

// Sine positional value for (rh, rw) inside an 8x8 window tile, channel c.
__device__ __forceinline__ float posval(int rh, int rw, int c) {
  float axis = (c < 64) ? (float)(rh + 1) : (float)(rw + 1);
  float v = axis / (8.f + 1e-6f) * 6.28318530718f;
  int cc = c & 63;
  float ang = v / __powf(10000.f, (float)((cc >> 1) * 2) / 64.f);
  return (cc & 1) ? __cosf(ang) : __sinf(ang);
}

// LN + sine position encoding (mutual-attention qkv input), f16 out.
__global__ void k_ln_pos(const float* __restrict__ X, const float* __restrict__ g,
                         const float* __restrict__ b, _Float16* __restrict__ O16,
                         int sh, int sw) {
  int t = blockIdx.x, c = threadIdx.x;
  float v = X[(size_t)t * CCH + c];
  __shared__ float red[128];
  red[c] = v; __syncthreads();
  for (int s = 64; s > 0; s >>= 1) { if (c < s) red[c] += red[c + s]; __syncthreads(); }
  float mean = red[0] / 128.f; __syncthreads();
  float d = v - mean;
  red[c] = d * d; __syncthreads();
  for (int s = 64; s > 0; s >>= 1) { if (c < s) red[c] += red[c + s]; __syncthreads(); }
  float var = red[0] / 128.f;
  float y = d * rsqrtf(var + 1e-5f) * g[c] + b[c];
  int h = (t >> 6) & 63, w = t & 63;
  int rh = ((h - sh + HH) % HH) & 7;
  int rw = ((w - sw + WWID) % WWID) & 7;
  O16[(size_t)t * CCH + c] = (_Float16)(y + posval(rh, rw, c));
}

// ---------------- Generic WMMA GEMM with TDM-staged weights ----------------
// O = A(MxK,f16) @ Bt^T (Bt is [N][K] f16) + bias + resid ; writes f32 and/or f16.
// Grids are always exact multiples (M%64==0, N%32==0, K%32==0).
// The 32xK f16 weight slab for this block is DMA'd to LDS by the Tensor Data
// Mover (wave 0 issues; TENSORcnt + barrier publish it), B fragments then come
// from LDS while A streams from global with prefetch.
__global__ void __launch_bounds__(256) k_gemm(
    const _Float16* __restrict__ A, const _Float16* __restrict__ Bt,
    const float* __restrict__ bias, const float* __restrict__ resid,
    float* __restrict__ O32, _Float16* __restrict__ O16, int M, int N, int K) {
  extern __shared__ _Float16 LB[];   // 32 * K halves
  const int lane = threadIdx.x & 31, wid = threadIdx.x >> 5;
  const int laneHi = lane >> 4, lcol = lane & 15;
  const int bm = blockIdx.x * 64 + (wid & 3) * 16;
  const int bnLocal = (wid >> 2) * 16;

  if (wid == 0) {
    tdm_load_2d_f16(0u, Bt + (size_t)blockIdx.y * 32 * K, K, 32);
    __builtin_amdgcn_s_wait_tensorcnt((short)0);
  }
  __syncthreads();

  v8f c = {};
  const _Float16* Ap = A + (size_t)(bm + lcol) * K;
  const _Float16* Bp = LB + (size_t)(bnLocal + lcol) * K + (laneHi ? 16 : 0);
  for (int k0 = 0; k0 < K; k0 += 32) {
    if (k0 + 32 < K) __builtin_prefetch((const void*)(Ap + k0 + 32), 0, 1);
    v16h a, b;
#pragma unroll
    for (int j = 0; j < 16; ++j) {
      a[j] = Ap[k0 + a_kmap(j, laneHi)];
      b[j] = Bp[k0 + j];
    }
    c = wmma16(a, b, c);
  }
#pragma unroll
  for (int r = 0; r < 8; ++r) {
    int row = bm + r + laneHi * 8;
    int col = blockIdx.y * 32 + bnLocal + lcol;
    float v = c[r];
    if (bias)  v += bias[col];
    if (resid) v += resid[(size_t)row * N + col];
    if (O32) O32[(size_t)row * N + col] = v;
    if (O16) O16[(size_t)row * N + col] = (_Float16)v;
  }
}

// ---------------- Window attention ----------------
__device__ __forceinline__ int wrap_tok(int rd, int rh, int rw, int sd, int sh, int sw) {
  int d = (rd + sd) % DD, h = (rh + sh) % HH, w = (rw + sw) % WWID;
  return (d * HH + h) * WWID + w;
}
__device__ __forceinline__ int regn(int p, int size, int ws, int ss) {
  if (ss == 0) return 0;
  return (p < size - ws) ? 0 : (p < size - ss) ? 1 : 2;
}

// One workgroup per (window, head [, dir]); 4 waves, each owns 16 query rows.
// QKV layout: [t][384] f16 (q | k | v, head-major 16 each).
__global__ void __launch_bounds__(128) k_attn(
    const _Float16* __restrict__ QKV, const float* __restrict__ rpb,
    _Float16* __restrict__ Out,
    int wd, int wh, int ww, int sd, int sh, int sw,
    int useMask, int mut, int outStride, int outOff) {
  extern __shared__ float smem[];
  const int lane = threadIdx.x & 31, wid = threadIdx.x >> 5;
  const int laneHi = lane >> 4, lcol = lane & 15;
  const int head = blockIdx.y;
  const int widx = blockIdx.x;
  const int dir = blockIdx.z;
  const int nWw = WWID / ww, nWh = HH / wh;
  const int iww = widx % nWw, iwh = (widx / nWw) % nWh, iwd = widx / (nWw * nWh);
  const int hwN = wh * ww;
  const int Nfull = wd * hwN;
  const int Nq = mut ? (Nfull >> 1) : Nfull;
  const int Nk = Nq;
  const int qd  = mut ? (dir == 0 ? 1 : 0) : 0;
  const int kvd = mut ? (dir == 0 ? 0 : 1) : 0;
  const int outd = mut ? dir : 0;
  float* S = smem + (size_t)wid * 16 * Nk;
  const float scale = 0.25f;  // dh = 16

  for (int qt = wid; qt * 16 < Nq; qt += 4) {
    // -------- A fragment: this lane's query row (scaled) --------
    int nq = qt * 16 + lcol;
    int ldq, lhq, lwq;
    if (mut) { ldq = qd; lhq = nq / ww; lwq = nq % ww; }
    else     { ldq = nq / hwN; int r = nq % hwN; lhq = r / ww; lwq = r % ww; }
    int tq = wrap_tok(iwd * wd + ldq, iwh * wh + lhq, iww * ww + lwq, sd, sh, sw);
    const _Float16* Qp = QKV + (size_t)tq * 384 + head * 16;
    v16h a;
#pragma unroll
    for (int j = 0; j < 16; ++j) {
      int k = a_kmap(j, laneHi);
      a[j] = (k < 16) ? (_Float16)(scale * (float)Qp[k]) : (_Float16)0.f;
    }
    // -------- S = Q K^T (dh=16 padded to K=32) --------
    for (int kt = 0; kt * 16 < Nk; ++kt) {
      int nk = kt * 16 + lcol;
      int ldk, lhk, lwk;
      if (mut) { ldk = kvd; lhk = nk / ww; lwk = nk % ww; }
      else     { ldk = nk / hwN; int r = nk % hwN; lhk = r / ww; lwk = r % ww; }
      int tk = wrap_tok(iwd * wd + ldk, iwh * wh + lhk, iww * ww + lwk, sd, sh, sw);
      const _Float16* Kp = QKV + (size_t)tk * 384 + 128 + head * 16;
      v16h b;
#pragma unroll
      for (int j = 0; j < 16; ++j) {
        int kk = j + (laneHi ? 16 : 0);
        b[j] = (kk < 16) ? Kp[kk] : (_Float16)0.f;
      }
      v8f c = {};
      c = wmma16(a, b, c);
#pragma unroll
      for (int r8 = 0; r8 < 8; ++r8) {
        int m  = r8 + laneHi * 8;      // query row in tile
        int np = qt * 16 + m;          // local query idx
        int nn = kt * 16 + lcol;       // local key idx
        float s = c[r8];
        int ldp, lhp, lwp, ldx, lhx, lwx;
        if (mut) { ldp = qd; lhp = np / ww; lwp = np % ww;
                   ldx = kvd; lhx = nn / ww; lwx = nn % ww; }
        else { ldp = np / hwN; int rr = np % hwN; lhp = rr / ww; lwp = rr % ww;
               ldx = nn / hwN; rr = nn % hwN;    lhx = rr / ww; lwx = rr % ww; }
        if (rpb) {
          int i0 = ldp - ldx + wd - 1;
          int i1 = lhp - lhx + wh - 1;
          int i2 = lwp - lwx + ww - 1;
          int idx = (i0 * (2 * wh - 1) + i1) * (2 * ww - 1) + i2;
          s += rpb[idx * 8 + head];
        }
        if (useMask) {
          int mldp = mut ? 0 : ldp, mldx = mut ? 0 : ldx;
          int cp = regn(iwd * wd + mldp, DD, wd, sd) * 9 +
                   regn(iwh * wh + lhp, HH, wh, sh) * 3 +
                   regn(iww * ww + lwp, WWID, ww, sw);
          int ck = regn(iwd * wd + mldx, DD, wd, sd) * 9 +
                   regn(iwh * wh + lhx, HH, wh, sh) * 3 +
                   regn(iww * ww + lwx, WWID, ww, sw);
          if (cp != ck) s -= 100.f;
        }
        S[m * Nk + nn] = s;
      }
    }
    // -------- row softmax (lane pair per row, halves combined via shfl) --------
    {
      int m = lcol, half = laneHi, Nk2 = Nk >> 1;
      float* row = S + m * Nk;
      int c0 = half * Nk2, c1 = c0 + Nk2;
      float mx = -3.0e38f;
      for (int cc = c0; cc < c1; ++cc) mx = fmaxf(mx, row[cc]);
      mx = fmaxf(mx, __shfl_xor(mx, 16));
      float sum = 0.f;
      for (int cc = c0; cc < c1; ++cc) { float e = __expf(row[cc] - mx); row[cc] = e; sum += e; }
      sum += __shfl_xor(sum, 16);
      float inv = 1.f / sum;
      for (int cc = c0; cc < c1; ++cc) row[cc] *= inv;
    }
    // -------- O = P @ V --------
    v8f o = {};
    for (int kt2 = 0; kt2 * 32 < Nk; ++kt2) {
      v16h a2, b2;
#pragma unroll
      for (int j = 0; j < 16; ++j) {
        int k = a_kmap(j, laneHi);
        a2[j] = (_Float16)S[lcol * Nk + kt2 * 32 + k];
        int kk = j + (laneHi ? 16 : 0);
        int nk = kt2 * 32 + kk;
        int ldk, lhk, lwk;
        if (mut) { ldk = kvd; lhk = nk / ww; lwk = nk % ww; }
        else     { ldk = nk / hwN; int rr = nk % hwN; lhk = rr / ww; lwk = rr % ww; }
        int tk = wrap_tok(iwd * wd + ldk, iwh * wh + lhk, iww * ww + lwk, sd, sh, sw);
        b2[j] = QKV[(size_t)tk * 384 + 256 + head * 16 + lcol];
      }
      o = wmma16(a2, b2, o);
    }
#pragma unroll
    for (int r8 = 0; r8 < 8; ++r8) {
      int m = r8 + laneHi * 8;
      int np = qt * 16 + m;
      int ldo, lho, lwo;
      if (mut) { ldo = outd; lho = np / ww; lwo = np % ww; }
      else     { ldo = np / hwN; int rr = np % hwN; lho = rr / ww; lwo = rr % ww; }
      int to = wrap_tok(iwd * wd + ldo, iwh * wh + lho, iww * ww + lwo, sd, sh, sw);
      Out[(size_t)to * outStride + outOff + head * 16 + lcol] = (_Float16)o[r8];
    }
  }
}

// ---------------- Elementwise helpers ----------------
__global__ void k_geglu(const float* __restrict__ h1, const float* __restrict__ h2,
                        _Float16* __restrict__ o, int n) {
  int i = blockIdx.x * 256 + threadIdx.x;
  if (i < n) {
    float x = h1[i];
    float gel = 0.5f * x * (1.f + erff(x * 0.70710678f));
    o[i] = (_Float16)(gel * h2[i]);
  }
}
__global__ void k_cvt(const float* __restrict__ in, _Float16* __restrict__ out, int n) {
  int i = blockIdx.x * 256 + threadIdx.x;
  if (i < n) out[i] = (_Float16)in[i];
}
__global__ void k_cvt_t(const float* __restrict__ in, _Float16* __restrict__ out, int K, int N) {
  int i = blockIdx.x * 256 + threadIdx.x;
  if (i < K * N) { int k = i / N, n = i % N; out[(size_t)n * K + k] = (_Float16)in[i]; }
}
__global__ void k_out_perm(const float* __restrict__ X, float* __restrict__ out) {
  int i = blockIdx.x * 256 + threadIdx.x;
  if (i < TOK * CCH) { int t = i >> 7, c = i & 127; out[(size_t)c * TOK + t] = X[(size_t)t * CCH + c]; }
}

// ---------------- Host orchestration ----------------
struct BlockP {
  const float *n1g, *n1b, *qkvw, *qkvb, *rpb, *projw, *projb;
  const float *n2g, *n2b, *fc11w, *fc11b, *fc12w, *fc12b, *fc2w, *fc2b;
  const float *qkvmw, *qkvmb;
};

extern "C" void kernel_launch(void* const* d_in, const int* in_sizes, int n_in,
                              void* d_out, int out_size, void* d_ws, size_t ws_size,
                              hipStream_t stream) {
  (void)in_sizes; (void)n_in; (void)out_size; (void)ws_size;
  auto F = [&](int i) { return (const float*)d_in[i]; };
  const float* x_in = F(0);
  const float* ln0g = F(1); const float* ln0b = F(2);
  const float* lin1w = F(3); const float* lin1b = F(4);
  const float* lin2w = F(5); const float* lin2b = F(6);
  auto blk = [&](int base, bool mut) {
    BlockP p;
    p.n1g = F(base + 0);  p.n1b = F(base + 1);
    p.qkvw = F(base + 2); p.qkvb = F(base + 3);
    p.rpb = F(base + 4);
    p.projw = F(base + 5); p.projb = F(base + 6);
    p.n2g = F(base + 7);  p.n2b = F(base + 8);
    p.fc11w = F(base + 9);  p.fc11b = F(base + 10);
    p.fc12w = F(base + 11); p.fc12b = F(base + 12);
    p.fc2w = F(base + 13);  p.fc2b = F(base + 14);
    p.qkvmw = mut ? F(base + 15) : nullptr;
    p.qkvmb = mut ? F(base + 16) : nullptr;
    return p;
  };
  BlockP g1[3] = { blk(7, true), blk(24, true), blk(41, true) };
  BlockP g2 = blk(58, false);

  // -------- workspace layout --------
  char* ws = (char*)d_ws;
  size_t off = 0;
  auto alloc = [&](size_t bytes) -> void* {
    void* p = ws + off; off = (off + bytes + 255) & ~(size_t)255; return p;
  };
  float*     X     = (float*)alloc((size_t)TOK * 128 * 4);
  float*     X0    = (float*)alloc((size_t)TOK * 128 * 4);
  _Float16*  LN16  = (_Float16*)alloc((size_t)TOK * 128 * 2);
  _Float16*  LNP16 = (_Float16*)alloc((size_t)TOK * 128 * 2);
  _Float16*  QKV16 = (_Float16*)alloc((size_t)TOK * 384 * 2);
  _Float16*  QKVM16= (_Float16*)alloc((size_t)TOK * 384 * 2);
  _Float16*  CAT16 = (_Float16*)alloc((size_t)TOK * 256 * 2);
  float*     H1    = (float*)alloc((size_t)TOK * 256 * 4);
  float*     H2    = (float*)alloc((size_t)TOK * 256 * 4);
  _Float16*  GE16  = (_Float16*)alloc((size_t)TOK * 256 * 2);
  _Float16*  Y16   = (_Float16*)alloc((size_t)TOK * 128 * 2);
  _Float16*  WT    = (_Float16*)alloc((size_t)49152 * 2);

  auto run_gemm = [&](const _Float16* A, const _Float16* Bt, const float* bias,
                      const float* resid, float* O32, _Float16* O16,
                      int M, int N, int K) {
    dim3 g(M / 64, N / 32);
    size_t smem = (size_t)32 * K * sizeof(_Float16);   // TDM-staged weight slab
    k_gemm<<<g, 256, smem, stream>>>(A, Bt, bias, resid, O32, O16, M, N, K);
  };
  auto cvt_t = [&](const float* w, int K, int N) {
    k_cvt_t<<<(K * N + 255) / 256, 256, 0, stream>>>(w, WT, K, N);
  };

  auto run_block = [&](const BlockP& p, int wd, int wh, int ww,
                       int sd, int sh, int sw, bool mut) {
    int shifted = (sd | sh | sw) ? 1 : 0;
    int nWin = (DD / wd) * (HH / wh) * (WWID / ww);
    int N = wd * wh * ww;
    int stride = mut ? 256 : 128;
    int mainOff = mut ? 128 : 0;
    // attention branch
    k_ln_tok<<<TOK, 128, 0, stream>>>(X, p.n1g, p.n1b, LN16, nullptr);
    cvt_t(p.qkvw, 128, 384);
    run_gemm(LN16, WT, p.qkvb, nullptr, nullptr, QKV16, TOK, 384, 128);
    size_t smem = (size_t)4 * 16 * N * 4;
    k_attn<<<dim3(nWin, 8, 1), 128, smem, stream>>>(
        QKV16, p.rpb, CAT16, wd, wh, ww, sd, sh, sw, shifted, 0, stride, mainOff);
    if (mut) {
      k_ln_pos<<<TOK, 128, 0, stream>>>(X, p.n1g, p.n1b, LNP16, sh, sw);
      cvt_t(p.qkvmw, 128, 384);
      run_gemm(LNP16, WT, p.qkvmb, nullptr, nullptr, QKVM16, TOK, 384, 128);
      size_t smem2 = (size_t)4 * 16 * (N / 2) * 4;
      k_attn<<<dim3(nWin, 8, 2), 128, smem2, stream>>>(
          QKVM16, nullptr, CAT16, wd, wh, ww, sd, sh, sw, shifted, 1, stride, 0);
    }
    int Kp = mut ? 256 : 128;
    cvt_t(p.projw, Kp, 128);
    run_gemm(CAT16, WT, p.projb, X, X, nullptr, TOK, 128, Kp);  // x += proj(attn)
    // GEGLU MLP
    k_ln_tok<<<TOK, 128, 0, stream>>>(X, p.n2g, p.n2b, LN16, nullptr);
    cvt_t(p.fc11w, 128, 256);
    run_gemm(LN16, WT, p.fc11b, nullptr, H1, nullptr, TOK, 256, 128);
    cvt_t(p.fc12w, 128, 256);
    run_gemm(LN16, WT, p.fc12b, nullptr, H2, nullptr, TOK, 256, 128);
    k_geglu<<<(TOK * 256 + 255) / 256, 256, 0, stream>>>(H1, H2, GE16, TOK * 256);
    cvt_t(p.fc2w, 256, 128);
    run_gemm(GE16, WT, p.fc2b, X, X, nullptr, TOK, 128, 256);   // x += mlp
  };

  // stage: ln0 + permute to token-major
  k_ln0<<<TOK, 128, 0, stream>>>(x_in, ln0g, ln0b, X);

  // residual group 1 (window 2x8x8, shift on odd blocks, mutual attention)
  (void)hipMemcpyAsync(X0, X, (size_t)TOK * 128 * 4, hipMemcpyDeviceToDevice, stream);
  run_block(g1[0], 2, 8, 8, 0, 0, 0, true);
  run_block(g1[1], 2, 8, 8, 1, 4, 4, true);
  run_block(g1[2], 2, 8, 8, 0, 0, 0, true);
  k_cvt<<<(TOK * 128 + 255) / 256, 256, 0, stream>>>(X, Y16, TOK * 128);
  cvt_t(lin1w, 128, 128);
  run_gemm(Y16, WT, lin1b, X0, X, nullptr, TOK, 128, 128);      // x = y@lin1 + b + x0

  // residual group 2 (window 6x8x8, single unshifted block)
  (void)hipMemcpyAsync(X0, X, (size_t)TOK * 128 * 4, hipMemcpyDeviceToDevice, stream);
  run_block(g2, 6, 8, 8, 0, 0, 0, false);
  k_cvt<<<(TOK * 128 + 255) / 256, 256, 0, stream>>>(X, Y16, TOK * 128);
  cvt_t(lin2w, 128, 128);
  run_gemm(Y16, WT, lin2b, X0, X, nullptr, TOK, 128, 128);      // x = y@lin2 + b + x0

  // back to (b, c, d, h, w)
  k_out_perm<<<(TOK * 128 + 255) / 256, 256, 0, stream>>>(X, (float*)d_out);
}
